// CompatibilityModel_66537633349983
// MI455X (gfx1250) — matrix operand order repaired and checked by reading, not compiled, not run on hardware
//
#include <hip/hip_runtime.h>
#include <hip/hip_bf16.h>

typedef __attribute__((ext_vector_type(16))) _Float16 v16h;
typedef __attribute__((ext_vector_type(8)))  _Float16 v8h;
typedef __attribute__((ext_vector_type(8)))  float    v8f;
typedef _Float16 half_t;

#define B_TOTAL (1 << 20)
#define BN_EPS 1e-5f

// ---------------------------------------------------------------------------
// Weight packing: f32 row-major [K x N] -> f16 WMMA B-fragments.
// B fragment layout used here (contiguous k, same as A): lane l holds col N=(l&15),
// K = ((l>=16)?16:0) + h for halfs h=0..15, within 32-K-step kTile.
// Fragments stored as wp[(fragId*32 + lane)*16 + h].
// Global frag ids: L1 -> 0..15 (2 kT x 8 nT), L2 -> 16..31 (4x4), L3 -> 32..35 (2x2)
// ---------------------------------------------------------------------------
__global__ void pack_weights(const float* __restrict__ W1,
                             const float* __restrict__ W2,
                             const float* __restrict__ W3,
                             half_t* __restrict__ wp) {
  int b = blockIdx.x;
  int lane = threadIdx.x;
  const float* W = W1;
  int Ksrc = 50, N = 128, nTiles = 8, fragBase = 0, f = b;
  if (b >= 32)      { W = W3; Ksrc = 64;  N = 32; nTiles = 2; fragBase = 32; f = b - 32; }
  else if (b >= 16) { W = W2; Ksrc = 128; N = 64; nTiles = 4; fragBase = 16; f = b - 16; }
  int kT = f / nTiles, nT = f % nTiles;
  int col = 16 * nT + (lane & 15);
  int kb  = 32 * kT + ((lane >= 16) ? 16 : 0);
  half_t* dst = wp + ((size_t)(fragBase + f) * 32 + lane) * 16;
#pragma unroll
  for (int h = 0; h < 16; ++h) {
    int K = kb + h;
    float v = (K < Ksrc) ? W[(size_t)K * N + col] : 0.0f;
    dst[h] = (half_t)v;
  }
}

// ---------------------------------------------------------------------------
// Fragment helpers
// ---------------------------------------------------------------------------
__device__ __forceinline__ v16h load_bfrag(const half_t* __restrict__ wp, int fragId, int lane) {
  return *(const v16h*)(wp + ((size_t)fragId * 32 + lane) * 16);
}

// A fragment from LDS (row-major x, stride in halfs). A 16x32 layout used here (contiguous k, same as B packing):
// lane l: row M=(l&15); kb=(l>=16)?16:0; halfs 0..15 -> K=kb+h
template <int STRIDE>
__device__ __forceinline__ v16h load_afrag(const half_t* xs, int rowBase, int kstep, int lane) {
  int row = rowBase + (lane & 15);
  int kb  = 32 * kstep + ((lane >= 16) ? 16 : 0);
  const half_t* p = xs + row * STRIDE + kb;
  v16h a;
#pragma unroll
  for (int h = 0; h < 8; ++h) { a[h] = p[h]; a[8 + h] = p[8 + h]; }
  return a;
}

__device__ __forceinline__ v8f wmma_f16(v16h a, v16h b, v8f c) {
  return __builtin_amdgcn_wmma_f32_16x16x32_f16(false, a, false, b, (short)0, c, false, false);
}

// ---------------------------------------------------------------------------
// C-tile writeback, tile-major: lane's 8 f32 -> 8 packed f16, single b128
// store at zg[(tileId*32 + lane)*8]. Also accumulates per-column stats.
// C layout: reg r, lane l: col N=(l&15), row M=r+((l>=16)?8:0).
// ---------------------------------------------------------------------------
__device__ __forceinline__ void store_tile_tm(v8f c, half_t* __restrict__ zg,
                                              size_t tileId, int lane, int colBase,
                                              float* ldsSum, float* ldsSq) {
  v8h hv;
  float s = 0.f, s2 = 0.f;
#pragma unroll
  for (int r = 0; r < 8; ++r) {
    float v = c[r];
    hv[r] = (half_t)v;
    s += v;
    s2 += v * v;
  }
  *(v8h*)(zg + (tileId * 32 + lane) * 8) = hv;
  int col = colBase + (lane & 15);
  atomicAdd(&ldsSum[col], s);
  atomicAdd(&ldsSq[col], s2);
}

// Row-major writeback (used for z3, consumed row-wise by final kernel)
__device__ __forceinline__ void store_tile_rm(v8f c, half_t* __restrict__ zg,
                                              int gRowBase, int Nld, int colBase,
                                              int lane, float* ldsSum, float* ldsSq) {
  int col  = colBase + (lane & 15);
  int rOff = (lane >= 16) ? 8 : 0;
  float s = 0.f, s2 = 0.f;
#pragma unroll
  for (int r = 0; r < 8; ++r) {
    float v = c[r];
    zg[(size_t)(gRowBase + rOff + r) * Nld + col] = (half_t)v;
    s += v;
    s2 += v * v;
  }
  atomicAdd(&ldsSum[col], s);
  atomicAdd(&ldsSq[col], s2);
}

// ---------------------------------------------------------------------------
// Feature encode (25 per pet). BN cancels b1..b3, so biases are skipped.
// ---------------------------------------------------------------------------
__device__ __forceinline__ void encode_one(half_t* x, int off, int breed, int size, int energy,
                                           int temp, float age, float soc, float wt,
                                           const float* be, const float* te) {
#pragma unroll
  for (int i = 0; i < 8; ++i) x[off + i] = (half_t)be[breed * 8 + i];
  x[off + 8]  = (half_t)(size == 0 ? 1.f : 0.f);
  x[off + 9]  = (half_t)(size == 1 ? 1.f : 0.f);
  x[off + 10] = (half_t)(size == 2 ? 1.f : 0.f);
  x[off + 11] = (half_t)(energy == 0 ? 1.f : 0.f);
  x[off + 12] = (half_t)(energy == 1 ? 1.f : 0.f);
  x[off + 13] = (half_t)(energy == 2 ? 1.f : 0.f);
#pragma unroll
  for (int i = 0; i < 8; ++i) x[off + 14 + i] = (half_t)te[temp * 8 + i];
  x[off + 22] = (half_t)(age * (1.0f / 15.0f));
  x[off + 23] = (half_t)soc;
  x[off + 24] = (half_t)(wt * (1.0f / 100.0f));
}

// ---------------------------------------------------------------------------
// Kernel 1: encode + layer1 GEMM (50->128, K padded to 64) + column stats.
// Output z1 in tile-major f16 layout.
// ---------------------------------------------------------------------------
__global__ void __launch_bounds__(256) encode_l1(
    const int* __restrict__ p1b, const int* __restrict__ p1s, const int* __restrict__ p1e,
    const int* __restrict__ p1t, const float* __restrict__ p1age, const float* __restrict__ p1soc,
    const float* __restrict__ p1w,
    const int* __restrict__ p2b, const int* __restrict__ p2s, const int* __restrict__ p2e,
    const int* __restrict__ p2t, const float* __restrict__ p2age, const float* __restrict__ p2soc,
    const float* __restrict__ p2w,
    const float* __restrict__ breed_emb, const float* __restrict__ temp_emb,
    const half_t* __restrict__ wp, half_t* __restrict__ z1,
    float* __restrict__ gSum, float* __restrict__ gSq) {
  __shared__ half_t xs[128][72];  // padded stride: 72 halfs avoids LDS bank conflicts
  __shared__ float be[15 * 8], te[9 * 8];
  __shared__ float sSum[128], sSq[128];
  int tid = threadIdx.x;
  if (tid < 120) be[tid] = breed_emb[tid];
  if (tid < 72)  te[tid] = temp_emb[tid];
  if (tid < 128) { sSum[tid] = 0.f; sSq[tid] = 0.f; }
  __syncthreads();

  if (tid < 128) {
    int row = blockIdx.x * 128 + tid;
    half_t* x = xs[tid];
    encode_one(x, 0,  p1b[row], p1s[row], p1e[row], p1t[row], p1age[row], p1soc[row], p1w[row], be, te);
    encode_one(x, 25, p2b[row], p2s[row], p2e[row], p2t[row], p2age[row], p2soc[row], p2w[row], be, te);
#pragma unroll
    for (int k = 50; k < 64; ++k) x[k] = (half_t)0.f;
  }
  __syncthreads();

  int wave = tid >> 5, lane = tid & 31;
  int rowBase = wave * 16;
  size_t rowTile = (size_t)blockIdx.x * 8 + wave;
  v16h a0 = load_afrag<72>(&xs[0][0], rowBase, 0, lane);
  v16h a1 = load_afrag<72>(&xs[0][0], rowBase, 1, lane);
#pragma unroll
  for (int nt = 0; nt < 8; ++nt) {
    v8f c = {};
    c = wmma_f16(a0, load_bfrag(wp, 0 * 8 + nt, lane), c);
    c = wmma_f16(a1, load_bfrag(wp, 1 * 8 + nt, lane), c);
    store_tile_tm(c, z1, rowTile * 8 + nt, lane, nt * 16, sSum, sSq);
  }
  __syncthreads();
  if (tid < 128) { atomicAdd(&gSum[tid], sSum[tid]); atomicAdd(&gSq[tid], sSq[tid]); }
}

// ---------------------------------------------------------------------------
// Fold BN stats + gamma/beta into per-column affine: y = z*scale + shift
// ---------------------------------------------------------------------------
__global__ void finalize_stats(const float* __restrict__ gSum, const float* __restrict__ gSq,
                               const float* __restrict__ gamma, const float* __restrict__ beta,
                               float* __restrict__ scale, float* __restrict__ shift, int n) {
  int j = threadIdx.x;
  if (j < n) {
    float invB = 1.0f / (float)B_TOTAL;
    float mu   = gSum[j] * invB;
    float var  = gSq[j] * invB - mu * mu;
    float rstd = rsqrtf(var + BN_EPS);
    float sc   = gamma[j] * rstd;
    scale[j] = sc;
    shift[j] = beta[j] - mu * sc;
  }
}

// ---------------------------------------------------------------------------
// Hidden layer: consume tile-major z, apply relu(z*scale+shift), GEMM
// (KIN -> NOUT) + column stats. Output tile-major (TM_OUT) or row-major.
// Each wave owns rowTile = blockIdx*8 + wave (its 16 rows of the 128-row block).
// ---------------------------------------------------------------------------
template <int KIN, int NOUT, int KT, int NT, bool TM_OUT>
__global__ void __launch_bounds__(256) hidden_layer(
    const half_t* __restrict__ zin, const float* __restrict__ scale,
    const float* __restrict__ shift, const half_t* __restrict__ wp, int fragBase,
    half_t* __restrict__ zout, float* __restrict__ gSum, float* __restrict__ gSq) {
  __shared__ half_t xs[128][KIN + 8];
  __shared__ float sc[KIN], sh[KIN];
  __shared__ float sSum[NOUT], sSq[NOUT];
  int tid = threadIdx.x;
  if (tid < KIN)  { sc[tid] = scale[tid]; sh[tid] = shift[tid]; }
  if (tid < NOUT) { sSum[tid] = 0.f; sSq[tid] = 0.f; }
  __syncthreads();

  int wave = tid >> 5, lane = tid & 31;
  size_t rowTile = (size_t)blockIdx.x * 8 + wave;
  int rBase = wave * 16 + ((lane >= 16) ? 8 : 0);
  constexpr int KTILES = KIN / 16;
#pragma unroll
  for (int nt = 0; nt < KTILES; ++nt) {
    v8h hv = *(const v8h*)(zin + ((rowTile * KTILES + nt) * 32 + lane) * 8);
    int col = nt * 16 + (lane & 15);
    float scv = sc[col], shv = sh[col];
#pragma unroll
    for (int r = 0; r < 8; ++r) {
      float v = (float)hv[r] * scv + shv;
      xs[rBase + r][col] = (half_t)fmaxf(v, 0.f);
    }
  }
  __syncthreads();

  int rowBase  = wave * 16;
  int gRowBase = blockIdx.x * 128 + rowBase;
  v16h a[KT];
#pragma unroll
  for (int ks = 0; ks < KT; ++ks) a[ks] = load_afrag<KIN + 8>(&xs[0][0], rowBase, ks, lane);
#pragma unroll
  for (int nt = 0; nt < NT; ++nt) {
    v8f c = {};
#pragma unroll
    for (int ks = 0; ks < KT; ++ks)
      c = wmma_f16(a[ks], load_bfrag(wp, fragBase + ks * NT + nt, lane), c);
    if constexpr (TM_OUT)
      store_tile_tm(c, zout, rowTile * NT + nt, lane, nt * 16, sSum, sSq);
    else
      store_tile_rm(c, zout, gRowBase, NOUT, nt * 16, lane, sSum, sSq);
  }
  __syncthreads();
  if (tid < NOUT) { atomicAdd(&gSum[tid], sSum[tid]); atomicAdd(&gSq[tid], sSq[tid]); }
}

// ---------------------------------------------------------------------------
// Final: relu(z3*scale+shift) . W4 + b4 -> sigmoid   (z3 is row-major)
// ---------------------------------------------------------------------------
__global__ void __launch_bounds__(256) final_out(
    const half_t* __restrict__ z3, const float* __restrict__ scale,
    const float* __restrict__ shift, const float* __restrict__ W4,
    const float* __restrict__ b4, float* __restrict__ out) {
  __shared__ float sc[32], sh[32], w[32];
  int tid = threadIdx.x;
  if (tid < 32) { sc[tid] = scale[tid]; sh[tid] = shift[tid]; w[tid] = W4[tid]; }
  __syncthreads();
  int row = blockIdx.x * 256 + tid;
  const half_t* zr = z3 + (size_t)row * 32;
  float acc = b4[0];
#pragma unroll
  for (int j = 0; j < 32; ++j) {
    float v = (float)zr[j] * sc[j] + sh[j];
    acc += fmaxf(v, 0.f) * w[j];
  }
  out[row] = 1.0f / (1.0f + __expf(-acc));
}

// ---------------------------------------------------------------------------
extern "C" void kernel_launch(void* const* d_in, const int* in_sizes, int n_in,
                              void* d_out, int out_size, void* d_ws, size_t ws_size,
                              hipStream_t stream) {
  (void)in_sizes; (void)n_in; (void)out_size; (void)ws_size;
  const int*   p1b  = (const int*)d_in[0];
  const int*   p1s  = (const int*)d_in[1];
  const int*   p1e  = (const int*)d_in[2];
  const int*   p1t  = (const int*)d_in[3];
  const float* p1a  = (const float*)d_in[4];
  const float* p1so = (const float*)d_in[5];
  const float* p1w  = (const float*)d_in[6];
  const int*   p2b  = (const int*)d_in[7];
  const int*   p2s  = (const int*)d_in[8];
  const int*   p2e  = (const int*)d_in[9];
  const int*   p2t  = (const int*)d_in[10];
  const float* p2a  = (const float*)d_in[11];
  const float* p2so = (const float*)d_in[12];
  const float* p2w  = (const float*)d_in[13];
  const float* bemb = (const float*)d_in[14];
  const float* temb = (const float*)d_in[15];
  const float* W1 = (const float*)d_in[16];
  const float* g1 = (const float*)d_in[18];
  const float* be1 = (const float*)d_in[19];
  const float* W2 = (const float*)d_in[20];
  const float* g2 = (const float*)d_in[22];
  const float* be2 = (const float*)d_in[23];
  const float* W3 = (const float*)d_in[24];
  const float* g3 = (const float*)d_in[26];
  const float* be3 = (const float*)d_in[27];
  const float* W4 = (const float*)d_in[28];
  const float* b4 = (const float*)d_in[29];

  char* ws = (char*)d_ws;
  size_t off = 0;
  half_t* z1 = (half_t*)(ws + off); off += (size_t)B_TOTAL * 128 * sizeof(half_t);
  half_t* z2 = (half_t*)(ws + off); off += (size_t)B_TOTAL * 64  * sizeof(half_t);
  half_t* z3 = (half_t*)(ws + off); off += (size_t)B_TOTAL * 32  * sizeof(half_t);
  half_t* wp = (half_t*)(ws + off); off += (size_t)36 * 32 * 16 * sizeof(half_t);
  float* acc = (float*)(ws + off);  off += 448 * sizeof(float);
  float* sums1 = acc;       float* sq1 = acc + 128;
  float* sums2 = acc + 256; float* sq2 = acc + 320;
  float* sums3 = acc + 384; float* sq3 = acc + 416;
  float* sc1 = (float*)(ws + off);
  float* sh1 = sc1 + 128;
  float* sc2 = sh1 + 128; float* sh2 = sc2 + 64;
  float* sc3 = sh2 + 64;  float* sh3 = sc3 + 32;

  hipMemsetAsync(acc, 0, 448 * sizeof(float), stream);
  pack_weights<<<36, 32, 0, stream>>>(W1, W2, W3, wp);

  encode_l1<<<B_TOTAL / 128, 256, 0, stream>>>(
      p1b, p1s, p1e, p1t, p1a, p1so, p1w,
      p2b, p2s, p2e, p2t, p2a, p2so, p2w,
      bemb, temb, wp, z1, sums1, sq1);
  finalize_stats<<<1, 128, 0, stream>>>(sums1, sq1, g1, be1, sc1, sh1, 128);

  hidden_layer<128, 64, 4, 4, true><<<B_TOTAL / 128, 256, 0, stream>>>(
      z1, sc1, sh1, wp, 16, z2, sums2, sq2);
  finalize_stats<<<1, 64, 0, stream>>>(sums2, sq2, g2, be2, sc2, sh2, 64);

  hidden_layer<64, 32, 2, 2, false><<<B_TOTAL / 128, 256, 0, stream>>>(
      z2, sc2, sh2, wp, 32, z3, sums3, sq3);
  finalize_stats<<<1, 32, 0, stream>>>(sums3, sq3, g3, be3, sc3, sh3, 32);

  final_out<<<B_TOTAL / 256, 256, 0, stream>>>(z3, sc3, sh3, W4, b4, (float*)d_out);
}
